// Attention_38079180046910
// MI455X (gfx1250) — matrix operand order repaired and checked by reading, not compiled, not run on hardware
//
#include <hip/hip_runtime.h>

// MI455X / gfx1250 attention block:
//   convert/transposed-weight prepass -> QKV GEMM -> RoPE -> flash attention -> O GEMM
// All matmuls via v_wmma_f32_16x16x32_bf16 (wave32); GEMM tiles staged to LDS with
// GLOBAL_LOAD_ASYNC_TO_LDS_B128 (ASYNCcnt) + double buffering when available.

typedef __bf16 bf16_t;
typedef __attribute__((ext_vector_type(16))) __bf16 v16bf;
typedef __attribute__((ext_vector_type(8)))  __bf16 v8bf;
typedef __attribute__((ext_vector_type(8)))  float  v8f;
typedef __attribute__((ext_vector_type(4)))  int    v4i;

#define LQ     2048
#define DMODEL 4096
#define NH     32
#define NKV    8
#define HD     128
#define QKV_N  6144     // H*HD + 2*KV*HD
#define NEGBIG (-1000000000.0f)
#define QSCALE 0.08838834764831845f  // 128^-0.5

#if defined(__AMDGCN__) && __has_builtin(__builtin_amdgcn_global_load_async_to_lds_b128)
#define HAVE_ASYNC 1
#else
#define HAVE_ASYNC 0
#endif

static __device__ __forceinline__ void wait_async_all() {
#if HAVE_ASYNC
#if __has_builtin(__builtin_amdgcn_s_wait_asynccnt)
    __builtin_amdgcn_s_wait_asynccnt(0);
#else
    asm volatile("s_wait_asynccnt 0x0" ::: "memory");
#endif
#endif
}

// Copy one 16-byte chunk global->LDS (async DMA when available).
static __device__ __forceinline__ void copy16(const bf16_t* gp, bf16_t* sp) {
#if HAVE_ASYNC
    __builtin_amdgcn_global_load_async_to_lds_b128(
        (v4i*)gp,
        (__attribute__((address_space(3))) v4i*)sp, 0, 0);
#else
    *(v8bf*)sp = *(const v8bf*)gp;
#endif
}

static __device__ __forceinline__ v8f vzero8() {
    v8f z;
#pragma unroll
    for (int i = 0; i < 8; ++i) z[i] = 0.0f;
    return z;
}

// Assemble a 16x32 bf16 A-fragment for one lane from two contiguous 8-element chunks.
// frag[j]   (j=0..7)  <- lo[j]  (K = 16*half + j)
// frag[8+j] (j=0..7)  <- hi[j]  (K = 16*half + 8 + j)
static __device__ __forceinline__ v16bf make_afrag(const bf16_t* lo, const bf16_t* hi) {
    v8bf a = *(const v8bf*)lo;
    v8bf b = *(const v8bf*)hi;
    return __builtin_shufflevector(a, b, 0,1,2,3,4,5,6,7,8,9,10,11,12,13,14,15);
}

static __device__ __forceinline__ v8f wmma_bf16(v16bf a, v16bf b, v8f c) {
    return __builtin_amdgcn_wmma_f32_16x16x32_bf16(false, a, false, b, (short)0, c, false, false);
}

// ---------------------------------------------------------------------------
// Prepass 1: f32 -> bf16 elementwise (for x).
// ---------------------------------------------------------------------------
__global__ __launch_bounds__(256) void convert_bf16_kernel(const float* __restrict__ in,
                                                           bf16_t* __restrict__ out, int n) {
    int i = blockIdx.x * 1024 + threadIdx.x * 4;
    if (i + 3 < n) {
#pragma unroll
        for (int j = 0; j < 4; ++j) out[i + j] = (bf16_t)in[i + j];
    }
}

// ---------------------------------------------------------------------------
// Prepass 2: transpose-convert weights: out[n][k] = bf16(in[k][n]); in is KxN f32.
// 32x32 tiles through padded LDS; 256 threads as (32 x 8).
// ---------------------------------------------------------------------------
__global__ __launch_bounds__(256) void transpose_convert_kernel(const float* __restrict__ in,
                                                                bf16_t* __restrict__ out,
                                                                int K, int N) {
    __shared__ float tile[32][33];
    const int tx = threadIdx.x & 31;
    const int ty = threadIdx.x >> 5;           // 0..7
    const int k0 = blockIdx.y * 32, n0 = blockIdx.x * 32;
#pragma unroll
    for (int i = 0; i < 4; ++i)
        tile[ty + i * 8][tx] = in[(size_t)(k0 + ty + i * 8) * N + n0 + tx];
    __syncthreads();
#pragma unroll
    for (int i = 0; i < 4; ++i)
        out[(size_t)(n0 + ty + i * 8) * K + k0 + tx] = (bf16_t)tile[tx][ty + i * 8];
}

// ---------------------------------------------------------------------------
// GEMM: C[M,N] = A[M,K] @ Bt[N,K]^T, all-bf16 inputs, f32 accumulate.
// WG = 256 threads (8 waves). WG tile 128x128; wave tile 64(M) x 32(N).
// Double-buffered LDS tiles filled by async global->LDS b128 copies.
// ---------------------------------------------------------------------------
template <bool OUT_BF16>
__global__ __launch_bounds__(256) void gemm_kernel(const bf16_t* __restrict__ A,
                                                   const bf16_t* __restrict__ Bt,
                                                   void* __restrict__ Cp,
                                                   int M, int N, int K) {
    __shared__ __align__(32) bf16_t As[2][128 * 40];  // [row][k], stride 40
    __shared__ __align__(32) bf16_t Bs[2][128 * 48];  // [n][k],   stride 48

    const int tid  = threadIdx.x;
    const int lane = tid & 31;
    const int l16  = lane & 15;   // row (A) / col (B,C)
    const int hl   = lane >> 4;   // lane half
    const int wave = tid >> 5;
    const int wm   = wave & 1;    // 2 wave rows
    const int wn   = wave >> 1;   // 4 wave cols
    const int m0   = blockIdx.y * 128;
    const int n0   = blockIdx.x * 128;

    const bf16_t* Ablk = A  + (size_t)m0 * K;
    const bf16_t* Bblk = Bt + (size_t)n0 * K;

    // Stage one 128x32 bf16 tile: 512 chunks of 16B, 2 per thread.
    auto stage = [&](const bf16_t* g, bf16_t* s, int ldss, int k0) {
#pragma unroll
        for (int i = 0; i < 2; ++i) {
            int idx = tid + i * 256;
            int row = idx >> 2, ch = idx & 3;
            copy16(g + (size_t)row * K + k0 + ch * 8, s + row * ldss + ch * 8);
        }
    };

    v8f acc[4][2];
#pragma unroll
    for (int mt = 0; mt < 4; ++mt)
#pragma unroll
        for (int nt = 0; nt < 2; ++nt) acc[mt][nt] = vzero8();

    stage(Ablk, As[0], 40, 0);
    stage(Bblk, Bs[0], 48, 0);
    wait_async_all();
    __syncthreads();

    int cur = 0;
    for (int k0 = 0; k0 < K; k0 += 32) {
        const int nxt = cur ^ 1;
        if (k0 + 32 < K) {          // prefetch next tiles while computing current
            stage(Ablk, As[nxt], 40, k0 + 32);
            stage(Bblk, Bs[nxt], 48, k0 + 32);
        }

        v16bf af[4];
#pragma unroll
        for (int mt = 0; mt < 4; ++mt) {
            const bf16_t* base = &As[cur][(wm * 64 + mt * 16 + l16) * 40];
            af[mt] = make_afrag(base + 16 * hl, base + 16 * hl + 8);
        }
#pragma unroll
        for (int nt = 0; nt < 2; ++nt) {
            v16bf bfr = *(const v16bf*)&Bs[cur][(wn * 32 + nt * 16 + l16) * 48 + 16 * hl];
#pragma unroll
            for (int mt = 0; mt < 4; ++mt)
                acc[mt][nt] = wmma_bf16(af[mt], bfr, acc[mt][nt]);
        }

        wait_async_all();
        __syncthreads();
        cur = nxt;
    }

#pragma unroll
    for (int mt = 0; mt < 4; ++mt)
#pragma unroll
        for (int nt = 0; nt < 2; ++nt)
#pragma unroll
            for (int j = 0; j < 8; ++j) {
                int row = m0 + wm * 64 + mt * 16 + 8 * hl + j;
                int col = n0 + wn * 32 + nt * 16 + l16;
                if (OUT_BF16) ((bf16_t*)Cp)[(size_t)row * N + col] = (bf16_t)acc[mt][nt][j];
                else          ((float*)Cp)[(size_t)row * N + col] = acc[mt][nt][j];
            }
}

// ---------------------------------------------------------------------------
// RoPE + scatter into attention layouts (qkv now bf16).
// q -> bf16 [H][L][HD] (scaled by 1/sqrt(HD)); k -> bf16 [KV][L][HD];
// v -> bf16 transposed [KV][HD][L].
// ---------------------------------------------------------------------------
__global__ __launch_bounds__(128) void rope_scatter_kernel(const bf16_t* __restrict__ qkv,
                                                           const float* __restrict__ cosb,
                                                           const float* __restrict__ sinb,
                                                           bf16_t* __restrict__ qb,
                                                           bf16_t* __restrict__ kb,
                                                           bf16_t* __restrict__ vt) {
    const int l    = blockIdx.x;
    const int head = blockIdx.y;   // 0..NH+2*NKV-1
    const int d    = threadIdx.x;  // 0..127
    const bf16_t* row = qkv + (size_t)l * QKV_N + head * HD;
    float x = (float)row[d];
    if (head < NH + NKV && d < 64) {
        float cs = cosb[l * 64 + d];
        float sn = sinb[l * 64 + d];
        float other = (d < 32) ? -(float)row[d + 32] : (float)row[d - 32];
        x = x * cs + other * sn;
    }
    if (head < NH) {
        qb[((size_t)head * LQ + l) * HD + d] = (bf16_t)(x * QSCALE);
    } else if (head < NH + NKV) {
        int kv = head - NH;
        kb[((size_t)kv * LQ + l) * HD + d] = (bf16_t)x;
    } else {
        int kv = head - NH - NKV;
        vt[((size_t)kv * HD + d) * LQ + l] = (bf16_t)x;
    }
}

// ---------------------------------------------------------------------------
// Flash attention: one wave = 16 query rows; WG = 4 waves = 64 rows of 1 head.
// Key blocks of 32: QK^T (8 WMMA) -> online softmax -> P via wave-private LDS
// -> PV (8 WMMA). Causal mask computed analytically.
// ---------------------------------------------------------------------------
__global__ __launch_bounds__(128) void attn_kernel(const bf16_t* __restrict__ qb,
                                                   const bf16_t* __restrict__ kbuf,
                                                   const bf16_t* __restrict__ vt,
                                                   bf16_t* __restrict__ ob) {
    __shared__ __align__(32) bf16_t Pl[4][16 * 40];  // per-wave P tile, stride 40

    const int tid  = threadIdx.x;
    const int lane = tid & 31;
    const int l16  = lane & 15;
    const int hl   = lane >> 4;
    const int wave = tid >> 5;
    const int head = blockIdx.y;
    const int kv   = head >> 2;                       // NH/NKV == 4
    const int q0   = (blockIdx.x * 4 + wave) * 16;

    const bf16_t* qrow  = qb   + (size_t)head * LQ * HD;
    const bf16_t* kbase = kbuf + (size_t)kv * LQ * HD;
    const bf16_t* vbase = vt   + (size_t)kv * HD * LQ;
    bf16_t* P = Pl[wave];

    v16bf qf[4];
#pragma unroll
    for (int kc = 0; kc < 4; ++kc) {
        const bf16_t* base = qrow + (size_t)(q0 + l16) * HD + kc * 32;
        qf[kc] = make_afrag(base + 16 * hl, base + 16 * hl + 8);
    }

    v8f o[8];
#pragma unroll
    for (int t = 0; t < 8; ++t) o[t] = vzero8();
    float mrow[8], lrow[8];
#pragma unroll
    for (int j = 0; j < 8; ++j) { mrow[j] = -1e30f; lrow[j] = 0.0f; }

    const int nblk = (q0 + 47) >> 5;   // ceil((q0+16)/32)
    for (int b = 0; b < nblk; ++b) {
        const int kb0 = b * 32;
        v8f s0 = vzero8(), s1 = vzero8();
#pragma unroll
        for (int kc = 0; kc < 4; ++kc) {
            v16bf b0 = *(const v16bf*)(kbase + (size_t)(kb0 + l16) * HD + kc * 32 + 16 * hl);
            v16bf b1 = *(const v16bf*)(kbase + (size_t)(kb0 + 16 + l16) * HD + kc * 32 + 16 * hl);
            s0 = wmma_bf16(qf[kc], b0, s0);
            s1 = wmma_bf16(qf[kc], b1, s1);
        }
        if (kb0 + 31 > q0) {   // causal mask on diagonal blocks (wave-uniform branch)
#pragma unroll
            for (int j = 0; j < 8; ++j) {
                int r = q0 + 8 * hl + j;
                s0[j] += (kb0 + l16      > r) ? NEGBIG : 0.0f;
                s1[j] += (kb0 + 16 + l16 > r) ? NEGBIG : 0.0f;
            }
        }
        // Online softmax; row j's 16 columns live across 16 lanes of this half.
#pragma unroll
        for (int j = 0; j < 8; ++j) {
            float t = fmaxf(s0[j], s1[j]);
            t = fmaxf(t, __shfl_xor(t, 1));
            t = fmaxf(t, __shfl_xor(t, 2));
            t = fmaxf(t, __shfl_xor(t, 4));
            t = fmaxf(t, __shfl_xor(t, 8));
            float mn = fmaxf(mrow[j], t);
            float alpha = __expf(mrow[j] - mn);
            mrow[j] = mn;
            float p0 = __expf(s0[j] - mn);
            float p1 = __expf(s1[j] - mn);
            s0[j] = p0; s1[j] = p1;
            float rs = p0 + p1;
            rs += __shfl_xor(rs, 1);
            rs += __shfl_xor(rs, 2);
            rs += __shfl_xor(rs, 4);
            rs += __shfl_xor(rs, 8);
            lrow[j] = lrow[j] * alpha + rs;
#pragma unroll
            for (int t2 = 0; t2 < 8; ++t2) o[t2][j] *= alpha;
        }
        // D-layout -> A-fragment layout via wave-private LDS (DS ops in-order per wave).
#pragma unroll
        for (int j = 0; j < 8; ++j) {
            P[(8 * hl + j) * 40 + l16]      = (bf16_t)s0[j];
            P[(8 * hl + j) * 40 + 16 + l16] = (bf16_t)s1[j];
        }
        const bf16_t* pr = &P[l16 * 40];
        v16bf pf = make_afrag(pr + 16 * hl, pr + 16 * hl + 8);
#pragma unroll
        for (int t2 = 0; t2 < 8; ++t2) {
            v16bf vf = *(const v16bf*)(vbase + (size_t)(t2 * 16 + l16) * LQ + kb0 + 16 * hl);
            o[t2] = wmma_bf16(pf, vf, o[t2]);
        }
    }

#pragma unroll
    for (int j = 0; j < 8; ++j) {
        float inv = 1.0f / lrow[j];
        int row = q0 + 8 * hl + j;
#pragma unroll
        for (int t2 = 0; t2 < 8; ++t2)
            ob[(size_t)row * DMODEL + head * HD + t2 * 16 + l16] = (bf16_t)(o[t2][j] * inv);
    }
}

// ---------------------------------------------------------------------------
extern "C" void kernel_launch(void* const* d_in, const int* in_sizes, int n_in,
                              void* d_out, int out_size, void* d_ws, size_t ws_size,
                              hipStream_t stream) {
    (void)in_sizes; (void)n_in; (void)out_size; (void)ws_size;
    const float* x     = (const float*)d_in[0];
    // d_in[1] = attention_mask (unused; causal mask computed analytically)
    const float* cosb  = (const float*)d_in[2];
    const float* sinb  = (const float*)d_in[3];
    const float* w_qkv = (const float*)d_in[4];
    const float* w_o   = (const float*)d_in[5];
    float* out = (float*)d_out;

    // Workspace layout (bf16, all offsets 32B-aligned):
    bf16_t* xb   = (bf16_t*)d_ws;                       // [L][D]        16.8 MB
    bf16_t* wT   = xb   + (size_t)LQ * DMODEL;          // [N][K] shared 50.3 MB (wqT then woT)
    bf16_t* qkvb = wT   + (size_t)QKV_N * DMODEL;       // [L][QKV_N]    25.2 MB (later reused as ob)
    bf16_t* qb   = qkvb + (size_t)LQ * QKV_N;           // [H][L][HD]    16.8 MB
    bf16_t* kb   = qb   + (size_t)NH  * LQ * HD;        // [KV][L][HD]    4.2 MB
    bf16_t* vt   = kb   + (size_t)NKV * LQ * HD;        // [KV][HD][L]    4.2 MB
    bf16_t* ob   = qkvb;                                // aliases dead qkv buffer

    // 0a) x -> bf16
    convert_bf16_kernel<<<(LQ * DMODEL) / 1024, 256, 0, stream>>>(x, xb, LQ * DMODEL);
    // 0b) w_qkv^T -> bf16 [QKV_N][DMODEL]
    transpose_convert_kernel<<<dim3(QKV_N / 32, DMODEL / 32), 256, 0, stream>>>(
        w_qkv, wT, DMODEL, QKV_N);
    // 1) qkv = x @ w_qkv  (bf16 out)
    gemm_kernel<true><<<dim3(QKV_N / 128, LQ / 128), 256, 0, stream>>>(
        xb, wT, qkvb, LQ, QKV_N, DMODEL);
    // 2) RoPE + scatter to q/k/v layouts
    rope_scatter_kernel<<<dim3(LQ, NH + 2 * NKV), 128, 0, stream>>>(
        qkvb, cosb, sinb, qb, kb, vt);
    // 3) flash attention (writes ob, aliasing the now-dead qkv buffer)
    attn_kernel<<<dim3(LQ / 64, NH), 128, 0, stream>>>(qb, kb, vt, ob);
    // 4) w_o^T -> bf16 [DMODEL][DMODEL] (wT free after step 1, stream-ordered)
    transpose_convert_kernel<<<dim3(DMODEL / 32, DMODEL / 32), 256, 0, stream>>>(
        w_o, wT, DMODEL, DMODEL);
    // 5) out = o @ w_o  (f32 out)
    gemm_kernel<false><<<dim3(DMODEL / 128, LQ / 128), 256, 0, stream>>>(
        ob, wT, out, LQ, DMODEL, NH * HD);
}
